// JBF_25202868093645
// MI455X (gfx1250) — matrix operand order repaired and checked by reading, not compiled, not run on hardware
//
#include <hip/hip_runtime.h>
#include <hip/hip_bf16.h>

// ---------------------------------------------------------------------------
// JBF network on MI455X (gfx1250), implicit-GEMM convolutions on WMMA f16.
//   source/guidance: [2,64,256,256] f32 NCHW
//   conv5x5+PReLU, conv5x5+PReLU, conv1x1 -> 200ch  (x2 branches)
//   ker = r*gd ; jbf = per-pixel 5x5 grouped bilateral tap-sum
//   conv3x3+PReLU, conv3x3, + source residual
// Data movement: NHWC f16 activations staged to LDS via
// GLOBAL_LOAD_ASYNC_TO_LDS_B128 (ASYNCcnt); LDS channel stride padded to
// 72 halves (36 banks) so 16-lane b128 B-fragment reads are conflict-free.
// Each wave computes a 32x32 output tile: 2 A-frags x 2 B-frags -> 4 WMMAs
// per 32-wide k-step (2 x 16B loads per WMMA).
// ---------------------------------------------------------------------------

typedef _Float16 h16;
typedef __attribute__((ext_vector_type(16))) _Float16 v16h;
typedef __attribute__((ext_vector_type(8)))  _Float16 v8h;
typedef __attribute__((ext_vector_type(8)))  float    v8f;

#define BN 2
#define CN 64
#define HN 256
#define WN 256
#define NPIX (BN * HN * WN)      // 131072
#define OCHEAD 256               // 200 padded to 256 (4 x 64)
#define OCREAL 200

// --------------------------- layout repack kernels -------------------------

// f32 NCHW -> f16 NHWC (C=64 innermost)
__global__ void k_pack_nhwc(const float* __restrict__ in, h16* __restrict__ out) {
  int idx = blockIdx.x * 256 + threadIdx.x;   // over NPIX*64, exact multiple
  int c   = idx & 63;
  int pix = idx >> 6;               // b*H*W + y*W + x
  int b   = pix >> 16;              // H*W = 65536
  int hw  = pix & 65535;
  out[idx] = (h16)in[(((size_t)b * CN + c) << 16) + hw];
}

// f32 OIHW -> f16 [OCpad][K], tap-major k = tap*64 + ci ; zero-pad rows o>=oc
__global__ void k_pack_w(const float* __restrict__ w, h16* __restrict__ out,
                         int oc, int K, int taps, int total) {
  int idx = blockIdx.x * 256 + threadIdx.x;   // over ocpad*K
  if (idx >= total) return;
  int o = idx / K;
  int k = idx - o * K;
  int tap = k >> 6;                 // IC == 64
  int ci  = k & 63;
  float v = (o < oc) ? w[((size_t)(o * 64 + ci)) * taps + tap] : 0.0f;
  out[idx] = (h16)v;
}

// ------------------------- implicit-GEMM conv (WMMA) -----------------------
// Block: 256 threads = 8 waves -> 64 outch x 128 pixel tile.
// Wave w: m-supertile = w&1 (32 outch), n-supertile = w>>1 (32 pixels).
// Per 32-wide k-step: 2 A frags + 2 B frags -> 4 WMMAs (16x16x32 f16).
template <int FS, bool ACT, bool FINAL>
__global__ __launch_bounds__(256, 2)
void k_conv_wmma(const h16* __restrict__ in16,        // NHWC f16 [B][H][W][64]
                 const h16* __restrict__ wt,          // [OCpad][K] tap-major f16
                 const float* __restrict__ bias, int oc_real,
                 const float* __restrict__ alpha_p,   // PReLU slope (scalar)
                 h16* __restrict__ out16, int ocpad,  // NHWC f16 out
                 float* __restrict__ outf,            // FINAL: f32 NCHW out
                 const float* __restrict__ resid)     // FINAL: residual (NCHW)
{
  constexpr int PAD     = (FS - 1) / 2;
  constexpr int PT      = 128;                         // pixels per block
  constexpr int TW      = PT + FS - 1;
  constexpr int K       = FS * FS * 64;
  constexpr int KSTEPS  = K / 32;
  constexpr int CSTRIDE = 72;   // halves per (row,col) cell: 36 banks, no conflicts

  __shared__ __align__(16) h16 smem[FS * TW * CSTRIDE];

  const int tid         = threadIdx.x;
  const int tilesPerRow = WN / PT;                     // 2
  const int tileId      = blockIdx.x;
  const int b   = tileId / (HN * tilesPerRow);
  const int rem = tileId - b * (HN * tilesPerRow);
  const int y0  = rem / tilesPerRow;
  const int x0  = (rem - y0 * tilesPerRow) * PT;

  // ---- stage input tile + halo into LDS as [row][col][ch(+pad)] -----------
  // In-bounds 16B chunks use the CDNA5 async-to-LDS engine (ASYNCcnt);
  // out-of-bounds halo chunks are zero-filled with plain ds stores (DScnt).
  const int chunks = FS * TW * 8;                      // 16B chunks
  for (int i = tid; i < chunks; i += 256) {
    int cchunk = i & 7;
    int pos    = i >> 3;
    int rr     = pos / TW;
    int cc     = pos - rr * TW;
    int y = y0 - PAD + rr;
    int x = x0 - PAD + cc;
    h16* dst = &smem[pos * CSTRIDE + (cchunk << 3)];
    if (y >= 0 && y < HN && x >= 0 && x < WN) {
      const h16* gp =
          &in16[((((size_t)b * HN + y) * WN + x) << 6) + (cchunk << 3)];
      unsigned ldsb = (unsigned)(size_t)dst;           // LDS byte address
      asm volatile("global_load_async_to_lds_b128 %0, %1, off"
                   :: "v"(ldsb), "v"(gp) : "memory");
    } else {
      v8h z;
      #pragma unroll
      for (int q = 0; q < 8; ++q) z[q] = (h16)0.0f;
      *(v8h*)dst = z;
    }
  }
  asm volatile("s_wait_asynccnt 0x0" ::: "memory");
  __syncthreads();

  const int wave   = tid >> 5;
  const int lane   = tid & 31;
  const int lane16 = lane & 15;
  const int hi     = lane >> 4;                        // 0 / 1 half-wave
  const int mt = wave & 1, nt = wave >> 1;
  const int mbase = blockIdx.y * 64 + mt * 32;
  const int nbase = nt * 32;
  const int n0 = nbase + lane16;                       // pixel frag 0
  const int n1 = n0 + 16;                              // pixel frag 1
  const int mrow0 = mbase + lane16;                    // A-matrix rows
  const int mrow1 = mrow0 + 16;

  v8f acc00 = {}, acc01 = {}, acc10 = {}, acc11 = {};
  // A fragment per-lane base: K offset kb = hi*8, second chunk at +16
  const h16* wrow0 = wt + (size_t)mrow0 * K + hi * 8;
  const h16* wrow1 = wt + (size_t)mrow1 * K + hi * 8;

  #pragma unroll 2
  for (int ks = 0; ks < KSTEPS; ++ks) {
    // A fragments: 16B global loads (contiguous K), each feeds two WMMAs
    v8h a0lo = *(const v8h*)(wrow0 + ks * 32);
    v8h a0hi = *(const v8h*)(wrow0 + ks * 32 + 16);
    v8h a1lo = *(const v8h*)(wrow1 + ks * 32);
    v8h a1hi = *(const v8h*)(wrow1 + ks * 32 + 16);
    __builtin_prefetch(wrow0 + (ks + 1) * 32, 0, 3);   // global_prefetch next
    __builtin_prefetch(wrow1 + (ks + 1) * 32, 0, 3);
    v16h a0, a1;
    #pragma unroll
    for (int q = 0; q < 8; ++q) {
      a0[q] = a0lo[q]; a0[q + 8] = a0hi[q];
      a1[q] = a1lo[q]; a1[q + 8] = a1hi[q];
    }

    // B fragments: per-lane K offset kb=hi*8, second chunk at +16 within this 32-chunk.
    // k = tap*64 + ci -> one tap per chunk, contiguous channels in LDS.
    int kglob = ks * 32 + hi * 8;
    int tap   = kglob >> 6;
    int ch0   = kglob & 63;
    int dy = tap / FS, dx = tap - dy * FS;
    const h16* bb0 = &smem[((dy * TW) + (n0 + dx)) * CSTRIDE + ch0];
    const h16* bb1 = &smem[((dy * TW) + (n1 + dx)) * CSTRIDE + ch0];
    v8h b0lo = *(const v8h*)bb0;
    v8h b0hi = *(const v8h*)(bb0 + 16);
    v8h b1lo = *(const v8h*)bb1;
    v8h b1hi = *(const v8h*)(bb1 + 16);
    v16h bf0, bf1;
    #pragma unroll
    for (int q = 0; q < 8; ++q) {
      bf0[q] = b0lo[q]; bf0[q + 8] = b0hi[q];
      bf1[q] = b1lo[q]; bf1[q + 8] = b1hi[q];
    }

    acc00 = __builtin_amdgcn_wmma_f32_16x16x32_f16(false, a0, false, bf0,
                                                   (short)0, acc00, false, false);
    acc01 = __builtin_amdgcn_wmma_f32_16x16x32_f16(false, a0, false, bf1,
                                                   (short)0, acc01, false, false);
    acc10 = __builtin_amdgcn_wmma_f32_16x16x32_f16(false, a1, false, bf0,
                                                   (short)0, acc10, false, false);
    acc11 = __builtin_amdgcn_wmma_f32_16x16x32_f16(false, a1, false, bf1,
                                                   (short)0, acc11, false, false);
  }

  // ---- epilogue: bias (+PReLU) -> f16 NHWC, or bias+residual -> f32 NCHW --
  float alpha = 0.0f;
  if constexpr (ACT) alpha = *alpha_p;

  #pragma unroll
  for (int mi = 0; mi < 2; ++mi) {
    const int moff = mbase + mi * 16 + hi * 8;         // 8 contiguous channels
    #pragma unroll
    for (int j = 0; j < 2; ++j) {
      v8f acc = mi ? (j ? acc11 : acc10) : (j ? acc01 : acc00);
      const int n  = j ? n1 : n0;
      const int px = x0 + n, py = y0;
      const size_t pix = ((size_t)b * HN + py) * WN + px;

      if constexpr (!FINAL) {
        v8h o8;
        #pragma unroll
        for (int v = 0; v < 8; ++v) {
          int ch  = moff + v;
          float o = acc[v] + ((ch < oc_real) ? bias[ch] : 0.0f);
          if constexpr (ACT) o = (o >= 0.0f) ? o : alpha * o;
          o8[v] = (h16)o;
        }
        *(v8h*)&out16[pix * (size_t)ocpad + moff] = o8;
      } else {
        #pragma unroll
        for (int v = 0; v < 8; ++v) {
          int ch    = moff + v;
          float o   = acc[v] + bias[ch];
          size_t oi = (((size_t)b * CN + ch) * HN + py) * WN + px;
          outf[oi] = o + resid[oi];
        }
      }
    }
  }
}

// ------------------------- per-pixel bilateral tap-sum ---------------------
// thread t -> pixel = t>>3, group g = t&7; 8 channels per group.
__global__ void k_jbf(const h16* __restrict__ src16,     // NHWC f16 [.,.,.,64]
                      const h16* __restrict__ rr,        // NHWC f16 [.,.,.,256]
                      const h16* __restrict__ gg,        // NHWC f16 [.,.,.,256]
                      h16* __restrict__ out16)           // NHWC f16 [.,.,.,64]
{
  int t   = blockIdx.x * 256 + threadIdx.x;    // NPIX*8, exact multiple
  int g   = t & 7;
  int pix = t >> 3;
  int b   = pix >> 16;
  int hw  = pix & 65535;
  int py = hw >> 8, px = hw & 255;

  float kv[25];
  const h16* rp = rr + (size_t)pix * OCHEAD + g * 25;
  const h16* gp = gg + (size_t)pix * OCHEAD + g * 25;
  #pragma unroll
  for (int k = 0; k < 25; ++k) kv[k] = (float)rp[k] * (float)gp[k];

  float acc[8];
  #pragma unroll
  for (int v = 0; v < 8; ++v) acc[v] = 0.0f;

  #pragma unroll
  for (int k = 0; k < 25; ++k) {
    int dy = k / 5 - 2, dx = k % 5 - 2;
    int y = py + dy, x = px + dx;
    if (y < 0 || y >= HN || x < 0 || x >= WN) continue;   // zero padding
    v8h s = *(const v8h*)&src16[((((size_t)b * HN + y) * WN + x) << 6) + g * 8];
    #pragma unroll
    for (int v = 0; v < 8; ++v) acc[v] += (float)s[v] * kv[k];
  }

  v8h o;
  #pragma unroll
  for (int v = 0; v < 8; ++v) o[v] = (h16)acc[v];
  *(v8h*)&out16[((size_t)pix << 6) + g * 8] = o;
}

// --------------------------------- driver ----------------------------------

extern "C" void kernel_launch(void* const* d_in, const int* in_sizes, int n_in,
                              void* d_out, int out_size, void* d_ws, size_t ws_size,
                              hipStream_t stream) {
  (void)in_sizes; (void)n_in; (void)out_size; (void)ws_size;

  const float* source   = (const float*)d_in[0];
  const float* guidance = (const float*)d_in[1];
  const float* t_w1 = (const float*)d_in[2];
  const float* t_b1 = (const float*)d_in[3];
  const float* t_a1 = (const float*)d_in[4];
  const float* t_w2 = (const float*)d_in[5];
  const float* t_b2 = (const float*)d_in[6];
  const float* t_a2 = (const float*)d_in[7];
  const float* t_w3 = (const float*)d_in[8];
  const float* t_b3 = (const float*)d_in[9];
  const float* g_w1 = (const float*)d_in[10];
  const float* g_b1 = (const float*)d_in[11];
  const float* g_a1 = (const float*)d_in[12];
  const float* g_w2 = (const float*)d_in[13];
  const float* g_b2 = (const float*)d_in[14];
  const float* g_a2 = (const float*)d_in[15];
  const float* g_w3 = (const float*)d_in[16];
  const float* g_b3 = (const float*)d_in[17];
  const float* j_w1 = (const float*)d_in[18];
  const float* j_b1 = (const float*)d_in[19];
  const float* j_a1 = (const float*)d_in[20];
  const float* j_w2 = (const float*)d_in[21];
  const float* j_b2 = (const float*)d_in[22];
  float* out = (float*)d_out;

  // ---- workspace layout (peak ~203 MB, buffers reused across phases) ----
  char* ws = (char*)d_ws;
  const size_t SZ64  = (size_t)NPIX * 64 * 2;        // 16 MB, NHWC f16 64ch
  const size_t SZ256 = (size_t)NPIX * OCHEAD * 2;    // 64 MB, NHWC f16 256ch
  h16* src16  = (h16*)(ws);
  h16* gsrc16 = (h16*)(ws + SZ64);
  h16* t1     = (h16*)(ws + 2 * SZ64);               // stage-1 act (both branches)
  h16* t2     = (h16*)(ws + 3 * SZ64);               // stage-2 act (both branches)
  h16* r256   = (h16*)(ws + 4 * SZ64);
  h16* g256   = (h16*)(ws + 4 * SZ64 + SZ256);
  char* wb    =        ws + 4 * SZ64 + 2 * SZ256;
  const size_t W5 = (size_t)64 * 1600 * 2;
  const size_t W1 = (size_t)OCHEAD * 64 * 2;
  const size_t W3 = (size_t)64 * 576 * 2;
  h16* wt1 = (h16*)wb;  wb += W5;
  h16* wt2 = (h16*)wb;  wb += W5;
  h16* wg1 = (h16*)wb;  wb += W5;
  h16* wg2 = (h16*)wb;  wb += W5;
  h16* wt3 = (h16*)wb;  wb += W1;
  h16* wg3 = (h16*)wb;  wb += W1;
  h16* wj1 = (h16*)wb;  wb += W3;
  h16* wj2 = (h16*)wb;  wb += W3;
  h16* jbf16 = t1;   // reuse: guidance stage-1 dead once g256 exists
  h16* y1    = t2;   // reuse: guidance stage-2 dead once g256 exists

  dim3 blk(256);

  // ---- phase 0: repack activations + weights to f16 ----
  k_pack_nhwc<<<NPIX * 64 / 256, blk, 0, stream>>>(source,   src16);
  k_pack_nhwc<<<NPIX * 64 / 256, blk, 0, stream>>>(guidance, gsrc16);

  k_pack_w<<<64 * 1600 / 256, blk, 0, stream>>>(t_w1, wt1, 64, 1600, 25, 64 * 1600);
  k_pack_w<<<64 * 1600 / 256, blk, 0, stream>>>(t_w2, wt2, 64, 1600, 25, 64 * 1600);
  k_pack_w<<<64 * 1600 / 256, blk, 0, stream>>>(g_w1, wg1, 64, 1600, 25, 64 * 1600);
  k_pack_w<<<64 * 1600 / 256, blk, 0, stream>>>(g_w2, wg2, 64, 1600, 25, 64 * 1600);
  k_pack_w<<<OCHEAD * 64 / 256, blk, 0, stream>>>(t_w3, wt3, OCREAL, 64, 1, OCHEAD * 64);
  k_pack_w<<<OCHEAD * 64 / 256, blk, 0, stream>>>(g_w3, wg3, OCREAL, 64, 1, OCHEAD * 64);
  k_pack_w<<<64 * 576 / 256, blk, 0, stream>>>(j_w1, wj1, 64, 576, 9, 64 * 576);
  k_pack_w<<<64 * 576 / 256, blk, 0, stream>>>(j_w2, wj2, 64, 576, 9, 64 * 576);

  const int PIXTILES = NPIX / 128;                   // 1024

  // ---- phase 1: target kernel-generator ----
  k_conv_wmma<5, true, false><<<dim3(PIXTILES, 1), blk, 0, stream>>>(
      src16, wt1, t_b1, 64, t_a1, t1, 64, nullptr, nullptr);
  k_conv_wmma<5, true, false><<<dim3(PIXTILES, 1), blk, 0, stream>>>(
      t1, wt2, t_b2, 64, t_a2, t2, 64, nullptr, nullptr);
  k_conv_wmma<1, false, false><<<dim3(PIXTILES, OCHEAD / 64), blk, 0, stream>>>(
      t2, wt3, t_b3, OCREAL, nullptr, r256, OCHEAD, nullptr, nullptr);

  // ---- phase 2: guidance kernel-generator (reuses t1/t2) ----
  k_conv_wmma<5, true, false><<<dim3(PIXTILES, 1), blk, 0, stream>>>(
      gsrc16, wg1, g_b1, 64, g_a1, t1, 64, nullptr, nullptr);
  k_conv_wmma<5, true, false><<<dim3(PIXTILES, 1), blk, 0, stream>>>(
      t1, wg2, g_b2, 64, g_a2, t2, 64, nullptr, nullptr);
  k_conv_wmma<1, false, false><<<dim3(PIXTILES, OCHEAD / 64), blk, 0, stream>>>(
      t2, wg3, g_b3, OCREAL, nullptr, g256, OCHEAD, nullptr, nullptr);

  // ---- phase 3: per-pixel grouped bilateral filtering ----
  k_jbf<<<NPIX * 8 / 256, blk, 0, stream>>>(src16, r256, g256, jbf16);

  // ---- phase 4: output convs + residual ----
  k_conv_wmma<3, true, false><<<dim3(PIXTILES, 1), blk, 0, stream>>>(
      jbf16, wj1, j_b1, 64, j_a1, y1, 64, nullptr, nullptr);
  k_conv_wmma<3, false, true><<<dim3(PIXTILES, 1), blk, 0, stream>>>(
      y1, wj2, j_b2, 64, nullptr, nullptr, 64, out, source);
}
